// ReactNet_75977971466569
// MI455X (gfx1250) — matrix operand order repaired and checked by reading, not compiled, not run on hardware
//
#include <hip/hip_runtime.h>
#include <math.h>

// ---------------- model dims ----------------
#define C_DIM   512
#define IN_DIMS 128
#define HID     256
#define NLAYERS 6
#define NEXP    5
#define KSIZE   31
#define INNER   512
#define SQ_DIM  64     // C/8
#define BATCH   8
#define TLEN    512
#define NTOK    (BATCH*TLEN)   // 4096

// ---------------- WMMA types ----------------
typedef __attribute__((ext_vector_type(2)))  __bf16 v2bf;
typedef __attribute__((ext_vector_type(4)))  __bf16 v4bf;
typedef __attribute__((ext_vector_type(8)))  __bf16 v8bf;
typedef __attribute__((ext_vector_type(16))) __bf16 v16bf;
typedef __attribute__((ext_vector_type(8)))  float  v8f;

// =====================================================================
// Tiled GEMM: C[M,N] (+)= act(A[M,K] @ B[K,N] + bias) * rowScale
// f32 in memory -> bf16 converted ONCE at LDS-store time,
// f32 accumulation via v_wmma_f32_16x16x32_bf16.
// Block tile 256x64, K-step 32, 256 threads = 8 waves.
// Each wave owns two 16-row strips (rows w*16 and w*16+128) x all 64 cols:
// 2 A fragments + 4 shared B fragments -> 8 independent WMMAs per K-step.
// All fragment slices are loaded into distinct registers BEFORE the WMMA
// burst so one s_wait_dscnt covers the whole group.
// LDS double-buffered; global loads for tile k+1 overlap WMMAs of tile k.
// LDS row stride 40 bf16 = 80B: 16B-aligned fragment slices (ds_load_b128),
// 20-dword bank stride -> conflict-free across 16 lanes.
// act: 0 none, 1 SiLU, 2 exact GELU.
// =====================================================================
#define BM 256
#define BN 64
#define BK 32
#define LDA 40
#define LDB 40

__global__ __launch_bounds__(256)
void gemm_wmma_bf16(const float* __restrict__ A, const float* __restrict__ B,
                    const float* __restrict__ bias,
                    const float* __restrict__ rowScale, int rsStride,
                    float* __restrict__ C,
                    int M, int N, int K, int act, int accumulate)
{
    __shared__ __align__(16) __bf16 As[2][BM][LDA];
    __shared__ __align__(16) __bf16 Bs[2][BN][LDB];   // transposed: [n][k]

    const int tid  = threadIdx.x;
    const int wave = tid >> 5;
    const int lane = tid & 31;
    const int m0 = blockIdx.x * BM;
    const int n0 = blockIdx.y * BN;

    v8f acc[8];   // [strip*4 + nt]
    for (int i = 0; i < 8; ++i)
        for (int j = 0; j < 8; ++j) acc[i][j] = 0.0f;

    // per-thread staging registers for one K-tile
    float4 aReg[8];   // A: 256x32 / 256 thr = 32 f32 = 8 x float4
    float2 bReg[4];   // B: 32x64  / 256 thr = 8 f32 = 4 K-pairs

    const int nk = (K + BK - 1) / BK;

    auto loadRegs = [&](int kt) {
        const int k0 = kt * BK;
        #pragma unroll
        for (int j = 0; j < 8; ++j) {
            int i  = tid + j * 256;
            int r  = i >> 3;              // / (BK/4)
            int cc = (i & 7) * 4;
            int gr = m0 + r;
            float4 v = make_float4(0.f, 0.f, 0.f, 0.f);
            if (gr < M && (k0 + cc + 3) < K)
                v = *reinterpret_cast<const float4*>(A + (size_t)gr * K + (k0 + cc));
            aReg[j] = v;
        }
        #pragma unroll
        for (int j = 0; j < 4; ++j) {
            int i   = tid + j * 256;
            int cc  = i & 63;
            int rp  = i >> 6;             // K-pair index 0..15
            int gr0 = k0 + rp * 2;
            int gc  = n0 + cc;
            float2 v = make_float2(0.f, 0.f);
            if (gc < N) {
                if (gr0     < K) v.x = B[(size_t)gr0       * N + gc];
                if (gr0 + 1 < K) v.y = B[(size_t)(gr0 + 1) * N + gc];
            }
            bReg[j] = v;
        }
    };

    auto storeRegs = [&](int buf) {
        #pragma unroll
        for (int j = 0; j < 8; ++j) {
            int i  = tid + j * 256;
            int r  = i >> 3;
            int cc = (i & 7) * 4;
            v4bf pk;
            pk[0] = (__bf16)aReg[j].x; pk[1] = (__bf16)aReg[j].y;
            pk[2] = (__bf16)aReg[j].z; pk[3] = (__bf16)aReg[j].w;
            *reinterpret_cast<v4bf*>(&As[buf][r][cc]) = pk;   // 8B store, 8B aligned
        }
        #pragma unroll
        for (int j = 0; j < 4; ++j) {
            int i  = tid + j * 256;
            int cc = i & 63;
            int rp = i >> 6;
            v2bf pk;
            pk[0] = (__bf16)bReg[j].x; pk[1] = (__bf16)bReg[j].y;
            *reinterpret_cast<v2bf*>(&Bs[buf][cc][rp * 2]) = pk;  // 4B store, transposed
        }
    };

    loadRegs(0);

    const int arow = wave * 16 + (lane & 15);
    const int ka0  = (lane < 16) ? 0 : 8;     // A: half-lanes K 0-7/8-15 then +16
    const int kb   = (lane < 16) ? 0 : 8;     // B: half-lanes K 0-7/8-15 then +16

    for (int kt = 0; kt < nk; ++kt) {
        const int buf = kt & 1;
        storeRegs(buf);
        __syncthreads();

        if (kt + 1 < nk) {
            loadRegs(kt + 1);
            // warm L2 one more tile ahead (global_prefetch_b8)
            if (kt + 2 < nk) {
                int pr = m0 + (tid & 255);
                if (pr < M)
                    __builtin_prefetch(A + (size_t)pr * K + (kt + 2) * BK, 0, 1);
                int br = (kt + 2) * BK + (tid & 31);
                if (br < K)
                    __builtin_prefetch(B + (size_t)br * N + n0, 0, 1);
            }
        }

        // ---- load ALL fragment slices into distinct registers first ----
        v8bf a0lo = *reinterpret_cast<const v8bf*>(&As[buf][arow][ka0]);
        v8bf a0hi = *reinterpret_cast<const v8bf*>(&As[buf][arow][ka0 + 16]);
        v8bf a1lo = *reinterpret_cast<const v8bf*>(&As[buf][arow + 128][ka0]);
        v8bf a1hi = *reinterpret_cast<const v8bf*>(&As[buf][arow + 128][ka0 + 16]);
        v8bf blo[4], bhi[4];
        #pragma unroll
        for (int nt = 0; nt < 4; ++nt) {
            const int col = nt * 16 + (lane & 15);
            blo[nt] = *reinterpret_cast<const v8bf*>(&Bs[buf][col][kb]);
            bhi[nt] = *reinterpret_cast<const v8bf*>(&Bs[buf][col][kb + 16]);
        }

        v16bf afrag0 = __builtin_shufflevector(a0lo, a0hi,
            0,1,2,3,4,5,6,7,8,9,10,11,12,13,14,15);
        v16bf afrag1 = __builtin_shufflevector(a1lo, a1hi,
            0,1,2,3,4,5,6,7,8,9,10,11,12,13,14,15);

        // ---- 8 independent WMMAs, shared B fragments across both strips ----
        #pragma unroll
        for (int nt = 0; nt < 4; ++nt) {
            v16bf bfrag = __builtin_shufflevector(blo[nt], bhi[nt],
                0,1,2,3,4,5,6,7,8,9,10,11,12,13,14,15);
            acc[nt]     = __builtin_amdgcn_wmma_f32_16x16x32_bf16(
                false, afrag0, false, bfrag, (short)0, acc[nt],     false, false);
            acc[4 + nt] = __builtin_amdgcn_wmma_f32_16x16x32_bf16(
                false, afrag1, false, bfrag, (short)0, acc[4 + nt], false, false);
        }
        // no trailing barrier: a buffer is only rewritten 2 iterations later,
        // after a barrier whose s_wait_dscnt retired all reads of it
    }

    // ---- epilogue: C/D layout: VGPR r -> M = r (lanes 0-15) / r+8 (lanes 16-31)
    const int coll = lane & 15;
    for (int s = 0; s < 2; ++s) {
        const int rowbase = m0 + wave * 16 + s * 128 + ((lane < 16) ? 0 : 8);
        for (int nt = 0; nt < 4; ++nt) {
            const int col = n0 + nt * 16 + coll;
            if (col >= N) continue;
            for (int r = 0; r < 8; ++r) {
                const int row = rowbase + r;
                if (row >= M) continue;
                float v = acc[s * 4 + nt][r];
                if (bias) v += bias[col];
                if (act == 1)      v = v / (1.0f + __expf(-v));                     // SiLU
                else if (act == 2) v = 0.5f * v * (1.0f + erff(v * 0.70710678f));   // GELU exact
                if (rowScale) v *= rowScale[(size_t)row * rsStride];
                const size_t idx = (size_t)row * N + col;
                if (accumulate) C[idx] += v; else C[idx] = v;
            }
        }
    }
}

// =====================================================================
// Elementwise / small kernels
// =====================================================================

// [B, M, T] -> [B*T, M]
__global__ void transpose_in_k(const float* __restrict__ src, float* __restrict__ dst,
                               int Mdim, long total)
{
    size_t idx = (size_t)blockIdx.x * blockDim.x + threadIdx.x;
    if (idx >= (size_t)total) return;
    int m = idx % Mdim;
    size_t n = idx / Mdim;
    int t = n % TLEN;
    int b = n / TLEN;
    dst[idx] = src[((size_t)b * Mdim + m) * TLEN + t];
}

// diffusion sinusoidal embedding base: [B, 512]
__global__ void diffemb_k(const float* __restrict__ step, float* __restrict__ emb)
{
    int idx = blockIdx.x * blockDim.x + threadIdx.x;
    const int half = C_DIM / 2;
    if (idx >= BATCH * C_DIM) return;
    int j = idx % C_DIM;
    int b = idx / C_DIM;
    int i = (j < half) ? j : j - half;
    float freq = __expf(-logf(10000.0f) * (float)i / (float)(half - 1));
    float v = step[b] * freq;
    emb[idx] = (j < half) ? sinf(v) : cosf(v);
}

// split x[N,1024] into x1/x2 [N,512] and add broadcast emb[B,1024]
__global__ void split_add_emb_k(const float* __restrict__ x, const float* __restrict__ emb,
                                float* __restrict__ x1, float* __restrict__ x2, long total)
{
    size_t idx = (size_t)blockIdx.x * blockDim.x + threadIdx.x;
    if (idx >= (size_t)total) return;
    int c = idx % C_DIM;
    size_t n = idx / C_DIM;
    int b = (int)(n / TLEN);
    x1[idx] = x[n * (2 * C_DIM) + c]         + emb[(size_t)b * (2 * C_DIM) + c];
    x2[idx] = x[n * (2 * C_DIM) + C_DIM + c] + emb[(size_t)b * (2 * C_DIM) + C_DIM + c];
}

// LayerNorm: one wave32 per token
__global__ __launch_bounds__(256)
void layernorm_k(const float* __restrict__ x, const float* __restrict__ g,
                 const float* __restrict__ b, float* __restrict__ out,
                 int Ntok, int Cd)
{
    int wave = threadIdx.x >> 5, lane = threadIdx.x & 31;
    int n = blockIdx.x * 8 + wave;
    if (n >= Ntok) return;
    const float* xp = x + (size_t)n * Cd;
    float s = 0.f, s2 = 0.f;
    for (int c = lane; c < Cd; c += 32) { float v = xp[c]; s += v; s2 += v * v; }
    for (int off = 16; off; off >>= 1) {
        s  += __shfl_xor(s,  off, 32);
        s2 += __shfl_xor(s2, off, 32);
    }
    float mean = s / Cd;
    float var  = s2 / Cd - mean * mean;
    float inv  = rsqrtf(var + 1e-5f);
    float* op = out + (size_t)n * Cd;
    for (int c = lane; c < Cd; c += 32) op[c] = (xp[c] - mean) * inv * g[c] + b[c];
}

// depthwise conv along T, K=31, SAME zero padding inside each batch
__global__ void dwconv_k(const float* __restrict__ h, const float* __restrict__ w,
                         const float* __restrict__ bias, float* __restrict__ out, long total)
{
    size_t idx = (size_t)blockIdx.x * blockDim.x + threadIdx.x;
    if (idx >= (size_t)total) return;
    int c = idx % C_DIM;
    size_t n = idx / C_DIM;
    int t = n % TLEN;
    int b = n / TLEN;
    float acc = bias[c];
    for (int k = 0; k < KSIZE; ++k) {
        int tt = t + k - KSIZE / 2;
        if (tt < 0 || tt >= TLEN) continue;
        acc += h[((size_t)b * TLEN + tt) * C_DIM + c] * w[k * C_DIM + c];
    }
    out[idx] = acc;
}

// top-2 softmax router weights from logits[N,5]
__global__ void router_topk_k(const float* __restrict__ logits, float* __restrict__ wts, int Ntok)
{
    int n = blockIdx.x * blockDim.x + threadIdx.x;
    if (n >= Ntok) return;
    float l[NEXP];
    for (int e = 0; e < NEXP; ++e) l[e] = logits[(size_t)n * NEXP + e];
    int i0 = 0;
    for (int e = 1; e < NEXP; ++e) if (l[e] > l[i0]) i0 = e;
    int i1 = -1;
    for (int e = 0; e < NEXP; ++e) {
        if (e == i0) continue;
        if (i1 < 0 || l[e] > l[i1]) i1 = e;
    }
    float e1 = __expf(l[i1] - l[i0]);
    float s = 1.0f + e1;
    for (int e = 0; e < NEXP; ++e) wts[(size_t)n * NEXP + e] = 0.0f;
    wts[(size_t)n * NEXP + i0] = 1.0f / s;
    wts[(size_t)n * NEXP + i1] = e1 / s;
}

// swiglu: raw[N,2I] -> out[N,I]
__global__ void swiglu_k(const float* __restrict__ raw, float* __restrict__ out, long total)
{
    size_t idx = (size_t)blockIdx.x * blockDim.x + threadIdx.x;
    if (idx >= (size_t)total) return;
    int i = idx % INNER;
    size_t n = idx / INNER;
    float a = raw[n * (2 * INNER) + i];
    float g = raw[n * (2 * INNER) + INNER + i];
    out[idx] = a * g / (1.0f + __expf(-g));
}

// pooled[b,c] = mean_t(h*senet_res + y)
__global__ void pool_k(const float* __restrict__ h, const float* __restrict__ senres,
                       const float* __restrict__ y, float* __restrict__ pooled)
{
    int idx = blockIdx.x * blockDim.x + threadIdx.x;
    if (idx >= BATCH * C_DIM) return;
    int c = idx % C_DIM, b = idx / C_DIM;
    float s = 0.f;
    for (int t = 0; t < TLEN; ++t) {
        size_t k = ((size_t)b * TLEN + t) * C_DIM + c;
        s += h[k] * senres[c] + y[k];
    }
    pooled[idx] = s / (float)TLEN;
}

// SE gate: gate[b,:] = sigmoid(silu(pooled@w1+b1)@w2+b2), one block per batch
__global__ __launch_bounds__(256)
void se_gate_k(const float* __restrict__ pooled,
               const float* __restrict__ w1, const float* __restrict__ b1,
               const float* __restrict__ w2, const float* __restrict__ b2,
               float* __restrict__ gate)
{
    __shared__ float pl[C_DIM];
    __shared__ float hid[SQ_DIM];
    int b = blockIdx.x;
    for (int c = threadIdx.x; c < C_DIM; c += 256) pl[c] = pooled[(size_t)b * C_DIM + c];
    __syncthreads();
    for (int j = threadIdx.x; j < SQ_DIM; j += 256) {
        float s = b1[j];
        for (int c = 0; c < C_DIM; ++c) s += pl[c] * w1[c * SQ_DIM + j];
        hid[j] = s / (1.0f + __expf(-s));
    }
    __syncthreads();
    for (int c = threadIdx.x; c < C_DIM; c += 256) {
        float s = b2[c];
        for (int j = 0; j < SQ_DIM; ++j) s += hid[j] * w2[j * C_DIM + c];
        gate[(size_t)b * C_DIM + c] = 1.0f / (1.0f + __expf(-s));
    }
}

// RevNet: x2 += x1 + y*gate[b,:]
__global__ void revnet_update_k(float* __restrict__ x2, const float* __restrict__ x1,
                                const float* __restrict__ y, const float* __restrict__ gate,
                                long total)
{
    size_t idx = (size_t)blockIdx.x * blockDim.x + threadIdx.x;
    if (idx >= (size_t)total) return;
    int c = idx % C_DIM;
    size_t n = idx / C_DIM;
    int b = (int)(n / TLEN);
    x2[idx] += x1[idx] + y[idx] * gate[(size_t)b * C_DIM + c];
}

// concat [x1|x2] -> cat[N,1024]
__global__ void concat_k(const float* __restrict__ x1, const float* __restrict__ x2,
                         float* __restrict__ cat, long total)
{
    size_t idx = (size_t)blockIdx.x * blockDim.x + threadIdx.x;
    if (idx >= (size_t)total) return;
    int c = idx % (2 * C_DIM);
    size_t n = idx / (2 * C_DIM);
    cat[idx] = (c < C_DIM) ? x1[n * C_DIM + c] : x2[n * C_DIM + (c - C_DIM)];
}

// proj[N,128] -> out[B,1,128,T]
__global__ void transpose_out_k(const float* __restrict__ proj, float* __restrict__ out, long total)
{
    size_t idx = (size_t)blockIdx.x * blockDim.x + threadIdx.x;
    if (idx >= (size_t)total) return;
    int t = idx % TLEN;
    int m = (idx / TLEN) % IN_DIMS;
    int b = idx / ((size_t)TLEN * IN_DIMS);
    out[idx] = proj[((size_t)b * TLEN + t) * IN_DIMS + m];
}

// =====================================================================
// Host orchestration
// =====================================================================
extern "C" void kernel_launch(void* const* d_in, const int* in_sizes, int n_in,
                              void* d_out, int out_size, void* d_ws, size_t ws_size,
                              hipStream_t stream)
{
    (void)in_sizes; (void)n_in; (void)out_size; (void)ws_size;
    auto P = [&](int i) -> const float* { return (const float*)d_in[i]; };

    // JAX pytree order: spec, diffusion_step, cond, then params (dict keys alphabetical)
    const float* spec   = P(0);
    const float* dstep  = P(1);
    const float* cond   = P(2);
    const float* cond_b = P(3);
    const float* cond_w = P(4);
    const float* de_b1  = P(5);
    const float* de_b2  = P(6);
    const float* de_w1  = P(7);
    const float* de_w2  = P(8);
    const float* in_b   = P(9);
    const float* in_w   = P(10);
    const int L0 = 11, LS = 19;   // layers: 6 x 19 leaves, keys alphabetical
    const float* norm_b = P(L0 + NLAYERS * LS + 0);
    const float* norm_g = P(L0 + NLAYERS * LS + 1);
    const float* out_b  = P(L0 + NLAYERS * LS + 2);
    const float* out_w  = P(L0 + NLAYERS * LS + 3);

    // ---- workspace carving ----
    char* ws = (char*)d_ws;
    size_t off = 0;
    auto alloc = [&](size_t nfloats) -> float* {
        float* p = (float*)(ws + off);
        off = (off + nfloats * sizeof(float) + 255) & ~(size_t)255;
        return p;
    };
    const size_t N = NTOK;
    float* specT  = alloc(N * IN_DIMS);
    float* condT  = alloc(N * HID);
    float* xbuf   = alloc(N * 2 * C_DIM);     // reused as cat after layers
    float* embb   = alloc(BATCH * C_DIM);
    float* embh   = alloc(BATCH * 4 * C_DIM);
    float* emb2   = alloc(BATCH * 2 * C_DIM);
    float* x1     = alloc(N * C_DIM);
    float* x2     = alloc(N * C_DIM);
    float* hbuf   = alloc(N * C_DIM);
    float* vbuf   = alloc(N * C_DIM);
    float* traw   = alloc(N * 2 * INNER);     // also holds router logits, final LN out
    float* h1     = alloc(N * INNER);         // also router hidden
    float* h2     = alloc(N * INNER);
    float* ybuf   = alloc(N * C_DIM);
    float* wts    = alloc(N * NEXP);
    float* pooled = alloc(BATCH * C_DIM);
    float* gate   = alloc(BATCH * C_DIM);
    float* proj   = alloc(N * IN_DIMS);
    float* rhid   = h1;
    float* logits = traw;
    float* catb   = xbuf;
    float* lnout  = traw;

    auto gemm = [&](const float* A, const float* Bm, const float* bias,
                    const float* rs, int rsStride, float* Cm,
                    int M, int Nn, int K, int act, int accum) {
        dim3 g((unsigned)((M + BM - 1) / BM), (unsigned)((Nn + BN - 1) / BN));
        gemm_wmma_bf16<<<g, 256, 0, stream>>>(A, Bm, bias, rs, rsStride, Cm,
                                              M, Nn, K, act, accum);
    };
    auto GRID = [](long total) { return (unsigned)((total + 255) / 256); };

    // ---- input + cond projections ----
    transpose_in_k<<<GRID((long)N * IN_DIMS), 256, 0, stream>>>(spec, specT, IN_DIMS, (long)N * IN_DIMS);
    transpose_in_k<<<GRID((long)N * HID), 256, 0, stream>>>(cond, condT, HID, (long)N * HID);
    gemm(specT, in_w,   in_b,   nullptr, 0, xbuf, N, 2 * C_DIM, IN_DIMS, 0, 0);
    gemm(condT, cond_w, cond_b, nullptr, 0, xbuf, N, 2 * C_DIM, HID,     0, 1);

    // ---- diffusion-step embedding MLP ----
    diffemb_k<<<GRID(BATCH * C_DIM), 256, 0, stream>>>(dstep, embb);
    gemm(embb, de_w1, de_b1, nullptr, 0, embh, BATCH, 4 * C_DIM, C_DIM,     2, 0); // GELU
    gemm(embh, de_w2, de_b2, nullptr, 0, emb2, BATCH, 2 * C_DIM, 4 * C_DIM, 0, 0);
    split_add_emb_k<<<GRID((long)N * C_DIM), 256, 0, stream>>>(xbuf, emb2, x1, x2, (long)N * C_DIM);

    // ---- RevNet layers ----
    for (int L = 0; L < NLAYERS; ++L) {
        const float* conv_b = P(L0 + L * LS + 0);
        const float* conv_w = P(L0 + L * LS + 1);
        const float* eb1    = P(L0 + L * LS + 2);
        const float* eb2    = P(L0 + L * LS + 3);
        const float* eb3    = P(L0 + L * LS + 4);
        const float* ew1    = P(L0 + L * LS + 5);
        const float* ew2    = P(L0 + L * LS + 6);
        const float* ew3    = P(L0 + L * LS + 7);
        const float* ln_b   = P(L0 + L * LS + 8);
        const float* ln_g   = P(L0 + L * LS + 9);
        const float* rb1    = P(L0 + L * LS + 10);
        const float* rb2    = P(L0 + L * LS + 11);
        const float* rw1    = P(L0 + L * LS + 12);
        const float* rw2    = P(L0 + L * LS + 13);
        const float* se_b1  = P(L0 + L * LS + 14);
        const float* se_b2  = P(L0 + L * LS + 15);
        const float* se_w1  = P(L0 + L * LS + 16);
        const float* se_w2  = P(L0 + L * LS + 17);
        const float* senres = P(L0 + L * LS + 18);

        layernorm_k<<<(N + 7) / 8, 256, 0, stream>>>(x1, ln_g, ln_b, hbuf, N, C_DIM);
        dwconv_k<<<GRID((long)N * C_DIM), 256, 0, stream>>>(hbuf, conv_w, conv_b, vbuf, (long)N * C_DIM);

        // router: silu(v@rw1+rb1)@rw2+rb2 -> top-2 softmax weights
        gemm(vbuf, rw1, rb1, nullptr, 0, rhid,   N, C_DIM, C_DIM, 1, 0);
        gemm(rhid, rw2, rb2, nullptr, 0, logits, N, NEXP,  C_DIM, 0, 0);
        router_topk_k<<<GRID(N), 256, 0, stream>>>(logits, wts, N);

        // y = v (residual base), then dense expert mix accumulated in gemm epilogue
        hipMemcpyAsync(ybuf, vbuf, (size_t)N * C_DIM * sizeof(float),
                       hipMemcpyDeviceToDevice, stream);
        for (int e = 0; e < NEXP; ++e) {
            const float* w1e = ew1 + (size_t)e * C_DIM * 2 * INNER;
            const float* w2e = ew2 + (size_t)e * INNER * 2 * INNER;
            const float* w3e = ew3 + (size_t)e * INNER * C_DIM;
            gemm(vbuf, w1e, eb1 + e * 2 * INNER, nullptr, 0, traw, N, 2 * INNER, C_DIM, 0, 0);
            swiglu_k<<<GRID((long)N * INNER), 256, 0, stream>>>(traw, h1, (long)N * INNER);
            gemm(h1, w2e, eb2 + e * 2 * INNER, nullptr, 0, traw, N, 2 * INNER, INNER, 0, 0);
            swiglu_k<<<GRID((long)N * INNER), 256, 0, stream>>>(traw, h2, (long)N * INNER);
            // y += wts[:,e] * (h2 @ ew3 + eb3)   (fused row-scale + accumulate)
            gemm(h2, w3e, eb3 + e * C_DIM, wts + e, NEXP, ybuf, N, C_DIM, INNER, 0, 1);
        }

        pool_k<<<GRID(BATCH * C_DIM), 256, 0, stream>>>(hbuf, senres, ybuf, pooled);
        se_gate_k<<<BATCH, 256, 0, stream>>>(pooled, se_w1, se_b1, se_w2, se_b2, gate);
        revnet_update_k<<<GRID((long)N * C_DIM), 256, 0, stream>>>(x2, x1, ybuf, gate, (long)N * C_DIM);

        float* tmp = x1; x1 = x2; x2 = tmp;   // RevNet swap
    }

    // ---- final norm + output projection ----
    concat_k<<<GRID((long)N * 2 * C_DIM), 256, 0, stream>>>(x1, x2, catb, (long)N * 2 * C_DIM);
    layernorm_k<<<(N + 7) / 8, 256, 0, stream>>>(catb, norm_g, norm_b, lnout, N, 2 * C_DIM);
    gemm(lnout, out_w, out_b, nullptr, 0, proj, N, IN_DIMS, 2 * C_DIM, 0, 0);
    transpose_out_k<<<GRID((long)BATCH * IN_DIMS * TLEN), 256, 0, stream>>>(
        proj, (float*)d_out, (long)BATCH * IN_DIMS * TLEN);
}
